// Local_Reconstruction_10436770529680
// MI455X (gfx1250) — matrix operand order repaired and checked by reading, not compiled, not run on hardware
//
#include <hip/hip_runtime.h>

typedef __attribute__((ext_vector_type(16))) __bf16 v16bf;
typedef __attribute__((ext_vector_type(8)))  float  v8f;

#define DIMC   256
#define NHEAD  8
#define HD     32
#define NT     64           // tokens per window
#define TOPK_  51
#define XS     256          // bf16 row stride (elems) for X/Q/K/OUT LDS buffers
#define VTS    64           // v-transposed row stride (tokens)
#define MODS   65           // f32 row stride for modulation buffer

union FragU { v16bf bf; uint4 u[2]; };

__device__ __forceinline__ unsigned short f2bf(float f) {
    unsigned u = __builtin_bit_cast(unsigned, f);
    unsigned r = u + 0x7FFFu + ((u >> 16) & 1u);   // round-to-nearest-even
    return (unsigned short)(r >> 16);
}

__device__ __forceinline__ v8f vzero() {
    v8f z;
#pragma unroll
    for (int i = 0; i < 8; ++i) z[i] = 0.0f;
    return z;
}

// A-matrix 16x32 bf16 fragment from row-major memory (rows = M, contiguous K).
// Lane m=lane&15 holds row m; lane half selects K {0..7,16..23} vs {8..15,24..31}.
__device__ __forceinline__ v16bf load_a_frag(const unsigned short* base, int stride, int lane) {
    int m  = lane & 15;
    int kh = (lane >> 4) & 1;
    const unsigned short* p = base + m * stride + kh * 8;
    FragU f;
    f.u[0] = *(const uint4*)(p);        // K = kh*8 + 0..7
    f.u[1] = *(const uint4*)(p + 16);   // K = 16 + kh*8 + 0..7
    return f.bf;
}

// B-matrix 32x16 bf16 fragment from row-major B^T (rows = N columns, contiguous K).
// Lane n=lane&15 holds column n; lane half selects K {0..7,16..23} vs {8..15,24..31}.
__device__ __forceinline__ v16bf load_b_frag(const unsigned short* base, int stride, int lane) {
    int n  = lane & 15;
    int kh = (lane >> 4) & 1;
    const unsigned short* p = base + n * stride + kh * 8;
    FragU f;
    f.u[0] = *(const uint4*)(p);
    f.u[1] = *(const uint4*)(p + 16);
    return f.bf;
}

__device__ __forceinline__ v8f wmma_bf16(v16bf a, v16bf b, v8f c) {
    return __builtin_amdgcn_wmma_f32_16x16x32_bf16(false, a, false, b, (short)0, c, false, false);
}

__global__ void convert_weights(const float* __restrict__ wqkv, const float* __restrict__ wproj,
                                unsigned short* __restrict__ wqkv_bf, unsigned short* __restrict__ wproj_bf) {
    int i = blockIdx.x * 256 + threadIdx.x;
    if (i < 768 * DIMC) wqkv_bf[i] = f2bf(wqkv[i]);
    if (i < DIMC * DIMC) wproj_bf[i] = f2bf(wproj[i]);
}

__global__ __launch_bounds__(256) void window_attn_kernel(
    const float* __restrict__ x, const float* __restrict__ var,
    const unsigned short* __restrict__ wqkv_bf, const float* __restrict__ qkv_b,
    const unsigned short* __restrict__ wproj_bf, const float* __restrict__ proj_b,
    const float* __restrict__ rpb_table, float* __restrict__ out)
{
    // LDS regions (total 154,912 B < 160 KB CU-mode budget)
    __shared__ __align__(16) unsigned short sX[NT * XS];     // x bf16; later attn heads 0..3
    __shared__ __align__(16) unsigned short sQ[NT * XS];     // var bf16 -> q bf16; later attn heads 4..7
    __shared__ __align__(16) unsigned short sK[NT * XS];     // k bf16; later pre-proj out bf16
    __shared__ __align__(16) unsigned short sVT[DIMC * VTS]; // v transposed: [feature][token]
    __shared__ float sMOD[NT * MODS];                        // var_self f32 -> modulation f32
    __shared__ float sRPB[225 * NHEAD];                      // relative position bias table

    const int tid  = threadIdx.x;
    const int wave = tid >> 5;
    const int lane = tid & 31;
    const int b    = blockIdx.x;
    const int n0   = lane & 15;
    const int hi   = lane >> 4;
    const float SCALE = 0.17677669529663687f;   // 32^-0.5

    // ---------------- Phase 0: stage x, var (bf16), rpb ----------------
    {
        const float4* xb = (const float4*)(x   + (size_t)b * NT * DIMC);
        const float4* vb = (const float4*)(var + (size_t)b * NT * DIMC);
#pragma unroll 4
        for (int i = tid; i < NT * DIMC / 4; i += 256) {
            float4 vx = xb[i];
            float4 vv = vb[i];
            int r = (i * 4) >> 8, c = (i * 4) & 255;
            unsigned x0 = (unsigned)f2bf(vx.x) | ((unsigned)f2bf(vx.y) << 16);
            unsigned x1 = (unsigned)f2bf(vx.z) | ((unsigned)f2bf(vx.w) << 16);
            unsigned v0 = (unsigned)f2bf(vv.x) | ((unsigned)f2bf(vv.y) << 16);
            unsigned v1 = (unsigned)f2bf(vv.z) | ((unsigned)f2bf(vv.w) << 16);
            *(unsigned*)&sX[r * XS + c]     = x0;
            *(unsigned*)&sX[r * XS + c + 2] = x1;
            *(unsigned*)&sQ[r * XS + c]     = v0;
            *(unsigned*)&sQ[r * XS + c + 2] = v1;
        }
        for (int i = tid; i < 225 * NHEAD; i += 256) sRPB[i] = rpb_table[i];
    }
    __syncthreads();

    // ---------------- Phase 1: var_self = var @ var^T (WMMA) ----------------
    // Tiles t=2*wave, 2*wave+1 share the same row tile ti -> load A once per k-step.
    {
        int ti  = (wave * 2) >> 2;
        int tj0 = (wave * 2) & 3;          // second tile is tj0+1 (never crosses ti boundary)
        v8f acc0 = vzero(), acc1 = vzero();
#pragma unroll
        for (int kk = 0; kk < 8; ++kk) {
            v16bf a  = load_a_frag(sQ + ti * 16 * XS + kk * 32, XS, lane);
            v16bf b0 = load_b_frag(sQ + tj0 * 16 * XS + kk * 32, XS, lane);
            v16bf b1 = load_b_frag(sQ + (tj0 + 1) * 16 * XS + kk * 32, XS, lane);
            acc0 = wmma_bf16(a, b0, acc0);
            acc1 = wmma_bf16(a, b1, acc1);
        }
#pragma unroll
        for (int r = 0; r < 8; ++r) {
            sMOD[(ti * 16 + hi * 8 + r) * MODS + tj0 * 16 + n0]       = acc0[r];
            sMOD[(ti * 16 + hi * 8 + r) * MODS + (tj0 + 1) * 16 + n0] = acc1[r];
        }
    }
    __syncthreads();

    // ---------------- Phase 1b: top-k rank -> modulation ----------------
    // rank = #{j : v_j > v_i}; ties have measure zero for continuous inputs.
    {
        int row = tid >> 2;
        int cbase = (tid & 3) * 16;
        float myv[16];
        int   rk[16];
#pragma unroll
        for (int c = 0; c < 16; ++c) { myv[c] = sMOD[row * MODS + cbase + c]; rk[c] = 0; }
        for (int j = 0; j < NT; ++j) {
            float vj = sMOD[row * MODS + j];
#pragma unroll
            for (int c = 0; c < 16; ++c)
                rk[c] += (vj > myv[c]) ? 1 : 0;
        }
        __syncthreads();
#pragma unroll
        for (int c = 0; c < 16; ++c)
            sMOD[row * MODS + cbase + c] = (rk[c] < TOPK_) ? 1.0f : 1.2f;
    }
    __syncthreads();

    // ---------------- Phase 2: qkv = x @ Wqkv^T + b (WMMA, bf16 weights) ----------------
#pragma unroll
    for (int pass = 0; pass < 2; ++pass) {
        int f0 = wave * 96 + pass * 48;
        v8f acc[4][3];
#pragma unroll
        for (int mi = 0; mi < 4; ++mi)
#pragma unroll
            for (int ni = 0; ni < 3; ++ni) acc[mi][ni] = vzero();
#pragma unroll 2
        for (int kk = 0; kk < 8; ++kk) {
            v16bf a[4], bf[3];
#pragma unroll
            for (int mi = 0; mi < 4; ++mi)
                a[mi] = load_a_frag(sX + mi * 16 * XS + kk * 32, XS, lane);
#pragma unroll
            for (int ni = 0; ni < 3; ++ni)
                bf[ni] = load_b_frag(wqkv_bf + (size_t)(f0 + ni * 16) * DIMC + kk * 32, DIMC, lane);
#pragma unroll
            for (int mi = 0; mi < 4; ++mi)
#pragma unroll
                for (int ni = 0; ni < 3; ++ni)
                    acc[mi][ni] = wmma_bf16(a[mi], bf[ni], acc[mi][ni]);
        }
#pragma unroll
        for (int ni = 0; ni < 3; ++ni) {
            int f = f0 + ni * 16 + n0;
            float bias = qkv_b[f];
#pragma unroll
            for (int mi = 0; mi < 4; ++mi)
#pragma unroll
                for (int r = 0; r < 8; ++r) {
                    int tok = mi * 16 + hi * 8 + r;
                    float val = acc[mi][ni][r] + bias;
                    if (f < 256) {                                       // q: fold in softmax scale
                        sQ[tok * XS + f] = f2bf(val * SCALE);
                    } else if (f < 512) {                                // k
                        sK[tok * XS + (f - 256)] = f2bf(val);
                    } else {                                             // v (transposed)
                        sVT[(f - 512) * VTS + tok] = f2bf(val);
                    }
                }
        }
    }
    __syncthreads();

    // ---------------- Phase 3: scores = (q_scaled @ k^T) * mod + rpb ; softmax ----------------
    const int h = wave;
    v8f sc[4][4];
#pragma unroll
    for (int i = 0; i < 4; ++i) {
        v16bf a = load_a_frag(sQ + i * 16 * XS + h * HD, XS, lane);
#pragma unroll
        for (int j = 0; j < 4; ++j) {
            v16bf bf = load_b_frag(sK + j * 16 * XS + h * HD, XS, lane);
            sc[i][j] = wmma_bf16(a, bf, vzero());
        }
    }
#pragma unroll
    for (int i = 0; i < 4; ++i)
#pragma unroll
        for (int r = 0; r < 8; ++r) {
            int nrow = i * 16 + hi * 8 + r;
            int hn = nrow >> 3, wn = nrow & 7;
#pragma unroll
            for (int j = 0; j < 4; ++j) {
                int m  = j * 16 + n0;
                int hm = m >> 3, wm = m & 7;
                int idx = (hn - hm + 7) * 15 + (wn - wm + 7);
                float mod = sMOD[nrow * MODS + m];
                sc[i][j][r] = sc[i][j][r] * mod + sRPB[idx * NHEAD + h];
            }
        }
    // row softmax: masks 1..8 reduce within each 16-lane half (one row each)
#pragma unroll
    for (int i = 0; i < 4; ++i)
#pragma unroll
        for (int r = 0; r < 8; ++r) {
            float mx = fmaxf(fmaxf(sc[i][0][r], sc[i][1][r]), fmaxf(sc[i][2][r], sc[i][3][r]));
#pragma unroll
            for (int d = 1; d < 16; d <<= 1) mx = fmaxf(mx, __shfl_xor(mx, d, 32));
            float s = 0.0f;
#pragma unroll
            for (int j = 0; j < 4; ++j) { sc[i][j][r] = __expf(sc[i][j][r] - mx); s += sc[i][j][r]; }
#pragma unroll
            for (int d = 1; d < 16; d <<= 1) s += __shfl_xor(s, d, 32);
            float inv = 1.0f / s;
#pragma unroll
            for (int j = 0; j < 4; ++j) sc[i][j][r] *= inv;
        }
    __syncthreads();   // q/x regions now reusable as attn tiles

    unsigned short* attnbuf = (h < 4) ? (sX + h * (NT * NT)) : (sQ + (h - 4) * (NT * NT));
#pragma unroll
    for (int i = 0; i < 4; ++i)
#pragma unroll
        for (int j = 0; j < 4; ++j)
#pragma unroll
            for (int r = 0; r < 8; ++r)
                attnbuf[(i * 16 + hi * 8 + r) * NT + j * 16 + n0] = f2bf(sc[i][j][r]);

    // ---------------- Phase 4: out_h = attn @ v (WMMA) ----------------
    v8f oa[4][2];
#pragma unroll
    for (int mi = 0; mi < 4; ++mi)
#pragma unroll
        for (int ni = 0; ni < 2; ++ni) oa[mi][ni] = vzero();
#pragma unroll
    for (int kk = 0; kk < 2; ++kk) {
        v16bf a[4];
#pragma unroll
        for (int mi = 0; mi < 4; ++mi)
            a[mi] = load_a_frag(attnbuf + mi * 16 * NT + kk * 32, NT, lane);
#pragma unroll
        for (int ni = 0; ni < 2; ++ni) {
            v16bf bf = load_b_frag(sVT + (h * HD + ni * 16) * VTS + kk * 32, VTS, lane);
#pragma unroll
            for (int mi = 0; mi < 4; ++mi)
                oa[mi][ni] = wmma_bf16(a[mi], bf, oa[mi][ni]);
        }
    }
#pragma unroll
    for (int mi = 0; mi < 4; ++mi)
#pragma unroll
        for (int ni = 0; ni < 2; ++ni)
#pragma unroll
            for (int r = 0; r < 8; ++r)
                sK[(mi * 16 + hi * 8 + r) * XS + h * HD + ni * 16 + n0] = f2bf(oa[mi][ni][r]);
    __syncthreads();

    // ---------------- Phase 5: final = out @ Wproj^T + b (WMMA) ----------------
    v8f po[4][2];
#pragma unroll
    for (int mi = 0; mi < 4; ++mi)
#pragma unroll
        for (int ni = 0; ni < 2; ++ni) po[mi][ni] = vzero();
#pragma unroll 2
    for (int kk = 0; kk < 8; ++kk) {
        v16bf a[4];
#pragma unroll
        for (int mi = 0; mi < 4; ++mi)
            a[mi] = load_a_frag(sK + mi * 16 * XS + kk * 32, XS, lane);
#pragma unroll
        for (int ni = 0; ni < 2; ++ni) {
            v16bf bf = load_b_frag(wproj_bf + (size_t)(wave * 32 + ni * 16) * DIMC + kk * 32, DIMC, lane);
#pragma unroll
            for (int mi = 0; mi < 4; ++mi)
                po[mi][ni] = wmma_bf16(a[mi], bf, po[mi][ni]);
        }
    }
    float* outb = out + (size_t)b * NT * DIMC;
#pragma unroll
    for (int ni = 0; ni < 2; ++ni) {
        int c = wave * 32 + ni * 16 + n0;
        float bias = proj_b[c];
#pragma unroll
        for (int mi = 0; mi < 4; ++mi)
#pragma unroll
            for (int r = 0; r < 8; ++r)
                outb[(mi * 16 + hi * 8 + r) * DIMC + c] = po[mi][ni][r] + bias;
    }
}

extern "C" void kernel_launch(void* const* d_in, const int* in_sizes, int n_in,
                              void* d_out, int out_size, void* d_ws, size_t ws_size,
                              hipStream_t stream) {
    const float* x      = (const float*)d_in[0];
    const float* var    = (const float*)d_in[1];
    const float* qkv_w  = (const float*)d_in[2];
    const float* qkv_b  = (const float*)d_in[3];
    const float* proj_w = (const float*)d_in[4];
    const float* proj_b = (const float*)d_in[5];
    const float* rpb    = (const float*)d_in[6];

    unsigned short* wqkv_bf  = (unsigned short*)d_ws;
    unsigned short* wproj_bf = wqkv_bf + 768 * DIMC;

    convert_weights<<<768, 256, 0, stream>>>(qkv_w, proj_w, wqkv_bf, wproj_bf);
    window_attn_kernel<<<2048, 256, 0, stream>>>(x, var, wqkv_bf, qkv_b,
                                                 wproj_bf, proj_b, rpb, (float*)d_out);
}
